// MPNN_29119878266940
// MI455X (gfx1250) — matrix operand order repaired and checked by reading, not compiled, not run on hardware
//
#include <hip/hip_runtime.h>

// MPNN layer for MI455X (gfx1250): f16 WMMA pipeline, f32 accumulate.
// h_E stream (614MB) read exactly once via async global->LDS double buffering;
// all GEMMs via v_wmma_f32_16x16x32_f16.

#define NNODES 20000
#define KNB    30
#define HDIM   128
#define EDIM   256
#define FDIM   512

typedef __attribute__((ext_vector_type(16))) _Float16 v16h;
typedef __attribute__((ext_vector_type(8)))  _Float16 v8h;
typedef __attribute__((ext_vector_type(4)))  _Float16 v4h;
typedef __attribute__((ext_vector_type(8)))  float    v8f;

// A-fragment (16x32 f16): per lane, halves 0..7 = K[k0..k0+7], halves 8..15 = K[k0+16..k0+23]
static __device__ inline v16h frag_a_f16(const _Float16* p) {
  v8h lo = *(const v8h*)p;
  v8h hi = *(const v8h*)(p + 16);
  return __builtin_shufflevector(lo, hi, 0,1,2,3,4,5,6,7,8,9,10,11,12,13,14,15);
}

static __device__ inline v16h frag_a_f32(const float* p) {
  v16h r;
#pragma unroll
  for (int i = 0; i < 8; ++i) r[i] = (_Float16)p[i];
#pragma unroll
  for (int i = 0; i < 8; ++i) r[i + 8] = (_Float16)p[16 + i];
  return r;
}

static __device__ inline v16h frag_b_f16(const _Float16* p, int ld) {
  v16h r;
#pragma unroll
  for (int i = 0; i < 8; ++i) r[i] = p[i * ld];
#pragma unroll
  for (int i = 0; i < 8; ++i) r[i + 8] = p[(16 + i) * ld];
  return r;
}
static __device__ inline v8f wmma16(v16h a, v16h b, v8f c) {
  return __builtin_amdgcn_wmma_f32_16x16x32_f16(false, a, false, b, (short)0, c, false, false);
}

// CDNA5 async DMA: global -> LDS, 16B per lane, tracked by ASYNCcnt (CDNA5 ISA).
// GVS addressing: mem = SGPR_base + VGPR_off32 + IOFFSET. VDST = LDS byte address.
static __device__ inline void async_load_b128(unsigned lds_addr, unsigned goff, const void* base) {
  asm volatile("global_load_async_to_lds_b128 %0, %1, %2"
               :: "v"(lds_addr), "v"(goff), "s"(base) : "memory");
}
static __device__ inline void wait_async0() {
  asm volatile("s_wait_asynccnt 0x0" ::: "memory");
}

// ---------------- fp32 -> fp16 weight conversion ----------------
__global__ void k_cvt(const float* __restrict__ s, _Float16* __restrict__ d, int n) {
  int i = blockIdx.x * 256 + threadIdx.x;
  if (i < n) d[i] = (_Float16)s[i];
}

// ---------------- v1 = h_V @ W1[:128,:] + b1   [N,128] ----------------
__global__ __launch_bounds__(256) void k_v1(const float* __restrict__ hV,
                                            const _Float16* __restrict__ w1h,
                                            const float* __restrict__ b1,
                                            float* __restrict__ v1) {
  const int lane = threadIdx.x & 31;
  const int w    = threadIdx.x >> 5;
  const int r    = lane & 15;
  const int hi   = lane >> 4;
  const int nb   = blockIdx.x * 16;
  const int col  = w * 16 + r;

  v8f acc;
#pragma unroll
  for (int i = 0; i < 8; ++i) acc[i] = 0.f;
  float bv = b1[col];

#pragma unroll
  for (int kt = 0; kt < 4; ++kt) {
    v16h a = frag_a_f32(hV + (size_t)(nb + r) * HDIM + kt * 32 + hi * 8);
    v16h b = frag_b_f16(w1h + (size_t)(kt * 32 + hi * 8) * HDIM + w * 16 + r, HDIM);
    acc = wmma16(a, b, acc);
  }
#pragma unroll
  for (int i = 0; i < 8; ++i)
    v1[(size_t)(nb + i + 8 * hi) * HDIM + col] = acc[i] + bv;
}

// ---------------- edge MLP + neighbor aggregation ----------------
__global__ __launch_bounds__(256) void k_edge_mlp(
    const float* __restrict__ hE,
    const float* __restrict__ v1,
    const _Float16* __restrict__ w1eh,  // [256][128] (W1 rows 128..383)
    const _Float16* __restrict__ w2h,   // [128][128]
    const _Float16* __restrict__ w3h,   // [128][128]
    const float* __restrict__ b2,
    const float* __restrict__ b3,
    float* __restrict__ dh,
    int nnodes, int gridStride) {
  __shared__ __align__(16) float    s_f32[2 * KNB * EDIM];  // 60KB: async-DMA double buffer
  __shared__ __align__(16) _Float16 s_hef[32 * EDIM];       // 16KB: f16 A-tile (rows 30,31 zero)
  __shared__ __align__(16) _Float16 s_m1[32 * HDIM];        // 8KB
  __shared__ __align__(16) _Float16 s_m2[32 * HDIM];        // 8KB
  __shared__ __align__(16) _Float16 s_w2[HDIM * HDIM];      // 32KB
  __shared__ __align__(16) _Float16 s_w3[HDIM * HDIM];      // 32KB

  const int tid  = threadIdx.x;
  const int lane = tid & 31;
  const int w    = tid >> 5;
  const int r    = lane & 15;
  const int hi   = lane >> 4;
  const int col  = w * 16 + r;

  // stage W2/W3 into LDS once (each 32KB, float4 = 8 halfs)
  for (int i = tid; i < HDIM * HDIM / 8; i += 256) {
    ((float4*)s_w2)[i] = ((const float4*)w2h)[i];
    ((float4*)s_w3)[i] = ((const float4*)w3h)[i];
  }
  // zero the two pad rows of the f16 A-tile once (never rewritten)
  for (int i = tid; i < 2 * EDIM / 8; i += 256)
    ((float4*)(s_hef + 30 * EDIM))[i] = make_float4(0.f, 0.f, 0.f, 0.f);

  const float b2v = b2[col];
  const float b3v = b3[col];

  const unsigned lds0 = (unsigned)(size_t)(void*)s_f32;
  int n = blockIdx.x;
  // prologue: kick off async stage of first node into buffer 0
  if (n < nnodes) {
    unsigned go = (unsigned)n * (KNB * EDIM * 4u);
    for (int i = tid; i < KNB * EDIM / 4; i += 256)
      async_load_b128(lds0 + i * 16u, go + i * 16u, hE);
  }
  int buf = 0;

  for (; n < nnodes; n += gridStride, buf ^= 1) {
    wait_async0();        // this wave's DMA into s_f32[buf] landed
    __syncthreads();      // ... and everyone else's; prior-iter compute done

    // kick off async stage of the NEXT node into the other buffer (overlaps compute)
    const int n2 = n + gridStride;
    if (n2 < nnodes) {
      unsigned go = (unsigned)n2 * (KNB * EDIM * 4u);
      unsigned lb = lds0 + (unsigned)(buf ^ 1) * (KNB * EDIM * 4u);
      for (int i = tid; i < KNB * EDIM / 4; i += 256)
        async_load_b128(lb + i * 16u, go + i * 16u, hE);
    }

    // convert staged fp32 tile -> shared f16 A-tile (once per block, not per wave)
    {
      const float4* sf = (const float4*)(s_f32 + buf * (KNB * EDIM));
      for (int i = tid; i < KNB * EDIM / 4; i += 256) {
        float4 x = sf[i];
        v4h h;
        h[0] = (_Float16)x.x; h[1] = (_Float16)x.y;
        h[2] = (_Float16)x.z; h[3] = (_Float16)x.w;
        *(v4h*)(s_hef + i * 4) = h;
      }
    }
    const float iv = v1[(size_t)n * HDIM + col];  // hides under the WMMA loop
    __syncthreads();

    // ---- layer 1: relu([32x256] @ W1E[256x128] + v1) ----
    v8f c0, c1;
#pragma unroll
    for (int i = 0; i < 8; ++i) { c0[i] = 0.f; c1[i] = 0.f; }
#pragma unroll
    for (int kt = 0; kt < 8; ++kt) {
      const int k0 = kt * 32 + hi * 8;
      v16h a0 = frag_a_f16(s_hef + r * EDIM + k0);
      v16h a1 = frag_a_f16(s_hef + (16 + r) * EDIM + k0);
      v16h b  = frag_b_f16(w1eh + (size_t)(kt * 32 + hi * 8) * HDIM + w * 16 + r, HDIM);
      c0 = wmma16(a0, b, c0);
      c1 = wmma16(a1, b, c1);
    }
#pragma unroll
    for (int i = 0; i < 8; ++i) {
      float x0 = c0[i] + iv; x0 = x0 > 0.f ? x0 : 0.f;
      float x1 = c1[i] + iv; x1 = x1 > 0.f ? x1 : 0.f;
      s_m1[(i + 8 * hi) * HDIM + col]      = (_Float16)x0;
      s_m1[(16 + i + 8 * hi) * HDIM + col] = (_Float16)x1;
    }
    __syncthreads();

    // ---- layer 2: relu(m1 @ W2 + b2), B from LDS ----
#pragma unroll
    for (int i = 0; i < 8; ++i) { c0[i] = 0.f; c1[i] = 0.f; }
#pragma unroll
    for (int kt = 0; kt < 4; ++kt) {
      const int k0 = kt * 32 + hi * 8;
      v16h a0 = frag_a_f16(s_m1 + r * HDIM + k0);
      v16h a1 = frag_a_f16(s_m1 + (16 + r) * HDIM + k0);
      v16h b  = frag_b_f16(s_w2 + (kt * 32 + hi * 8) * HDIM + w * 16 + r, HDIM);
      c0 = wmma16(a0, b, c0);
      c1 = wmma16(a1, b, c1);
    }
#pragma unroll
    for (int i = 0; i < 8; ++i) {
      float x0 = c0[i] + b2v; x0 = x0 > 0.f ? x0 : 0.f;
      float x1 = c1[i] + b2v; x1 = x1 > 0.f ? x1 : 0.f;
      s_m2[(i + 8 * hi) * HDIM + col]      = (_Float16)x0;
      s_m2[(16 + i + 8 * hi) * HDIM + col] = (_Float16)x1;
    }
    __syncthreads();

    // ---- layer 3: m2 @ W3 + b3, sum the 30 valid rows ----
#pragma unroll
    for (int i = 0; i < 8; ++i) { c0[i] = 0.f; c1[i] = 0.f; }
#pragma unroll
    for (int kt = 0; kt < 4; ++kt) {
      const int k0 = kt * 32 + hi * 8;
      v16h a0 = frag_a_f16(s_m2 + r * HDIM + k0);
      v16h a1 = frag_a_f16(s_m2 + (16 + r) * HDIM + k0);
      v16h b  = frag_b_f16(s_w3 + (kt * 32 + hi * 8) * HDIM + w * 16 + r, HDIM);
      c0 = wmma16(a0, b, c0);
      c1 = wmma16(a1, b, c1);
    }
    float s = 0.f;
#pragma unroll
    for (int i = 0; i < 8; ++i) s += c0[i] + b3v;
    const int lim = (hi == 0) ? 8 : 6;  // drop padded rows 30 (hi=1,i=6) and 31 (hi=1,i=7)
#pragma unroll
    for (int i = 0; i < 8; ++i) if (i < lim) s += c1[i] + b3v;
    s += __shfl_xor(s, 16, 32);         // lanes l and l+16 share a column
    if (lane < 16)
      dh[(size_t)n * HDIM + w * 16 + lane] = s * (1.0f / 30.0f);
  }
}

// ---------------- instnorm -> FFN -> instnorm -> per-block column sums ----------------
__global__ __launch_bounds__(256) void k_ffn(
    const float* __restrict__ hV,
    const float* __restrict__ dh,
    const _Float16* __restrict__ wd1h,  // [128][512]
    const float* __restrict__ bd1,
    const _Float16* __restrict__ wd2h,  // [512][128]
    const float* __restrict__ bd2,
    float* __restrict__ partials) {
  __shared__ __align__(16) float    s_x[32 * HDIM];        // 16KB residual (normalized h)
  __shared__ __align__(16) _Float16 s_h[32 * HDIM];        // 8KB f16 A operand
  __shared__ __align__(16) char     s_buf[32 * FDIM * 2];  // 32KB: f16 y, later aliased f32 x2

  _Float16* s_y  = (_Float16*)s_buf;
  float*    s_x2 = (float*)s_buf;

  const int tid  = threadIdx.x;
  const int lane = tid & 31;
  const int w    = tid >> 5;
  const int rr   = lane & 15;
  const int hi   = lane >> 4;
  const int nb   = blockIdx.x * 32;

  // ---- norm1: h = instnorm(hV + dh); 8 threads per row, 16 elems each ----
  {
    const int row = tid >> 3;
    const int sub = tid & 7;
    float v[16];
    float sum = 0.f, sq = 0.f;
#pragma unroll
    for (int j = 0; j < 16; ++j) {
      int c = sub * 16 + j;
      float x = hV[(size_t)(nb + row) * HDIM + c] + dh[(size_t)(nb + row) * HDIM + c];
      v[j] = x; sum += x; sq += x * x;
    }
#pragma unroll
    for (int m = 1; m < 8; m <<= 1) { sum += __shfl_xor(sum, m, 32); sq += __shfl_xor(sq, m, 32); }
    float mu   = sum * (1.f / HDIM);
    float rstd = rsqrtf(sq * (1.f / HDIM) - mu * mu + 1e-5f);
#pragma unroll
    for (int j = 0; j < 16; ++j) {
      int c = sub * 16 + j;
      float xn = (v[j] - mu) * rstd;
      s_x[row * HDIM + c] = xn;
      s_h[row * HDIM + c] = (_Float16)xn;
    }
  }
  __syncthreads();

  // ---- GEMM1: y = relu(h @ Wd1 + bd1), [32x128]x[128x512]; wave w owns 4 N-tiles ----
#pragma unroll
  for (int nt4 = 0; nt4 < 4; ++nt4) {
    const int nt  = w * 4 + nt4;
    const int col = nt * 16 + rr;
    float bv = bd1[col];
    v8f c0, c1;
#pragma unroll
    for (int i = 0; i < 8; ++i) { c0[i] = 0.f; c1[i] = 0.f; }
#pragma unroll
    for (int kt = 0; kt < 4; ++kt) {
      const int k0 = kt * 32 + hi * 8;
      v16h a0 = frag_a_f16(s_h + rr * HDIM + k0);
      v16h a1 = frag_a_f16(s_h + (16 + rr) * HDIM + k0);
      v16h b  = frag_b_f16(wd1h + (size_t)(kt * 32 + hi * 8) * FDIM + nt * 16 + rr, FDIM);
      c0 = wmma16(a0, b, c0);
      c1 = wmma16(a1, b, c1);
    }
#pragma unroll
    for (int i = 0; i < 8; ++i) {
      float x0 = c0[i] + bv; x0 = x0 > 0.f ? x0 : 0.f;
      float x1 = c1[i] + bv; x1 = x1 > 0.f ? x1 : 0.f;
      s_y[(i + 8 * hi) * FDIM + col]      = (_Float16)x0;
      s_y[(16 + i + 8 * hi) * FDIM + col] = (_Float16)x1;
    }
  }
  __syncthreads();

  // ---- GEMM2: x2 = h + (y @ Wd2 + bd2) ----
  {
    const int col = w * 16 + rr;
    float bv = bd2[col];
    v8f c0, c1;
#pragma unroll
    for (int i = 0; i < 8; ++i) { c0[i] = 0.f; c1[i] = 0.f; }
#pragma unroll
    for (int kt = 0; kt < 16; ++kt) {
      const int k0 = kt * 32 + hi * 8;
      v16h a0 = frag_a_f16(s_y + rr * FDIM + k0);
      v16h a1 = frag_a_f16(s_y + (16 + rr) * FDIM + k0);
      v16h b  = frag_b_f16(wd2h + (size_t)(kt * 32 + hi * 8) * HDIM + w * 16 + rr, HDIM);
      c0 = wmma16(a0, b, c0);
      c1 = wmma16(a1, b, c1);
    }
    float x2v[16];
#pragma unroll
    for (int i = 0; i < 8; ++i) {
      x2v[i]     = s_x[(i + 8 * hi) * HDIM + col] + c0[i] + bv;
      x2v[8 + i] = s_x[(16 + i + 8 * hi) * HDIM + col] + c1[i] + bv;
    }
    __syncthreads();  // all reads of s_y / s_x complete before aliasing s_buf as f32
#pragma unroll
    for (int i = 0; i < 8; ++i) {
      s_x2[(i + 8 * hi) * HDIM + col]      = x2v[i];
      s_x2[(16 + i + 8 * hi) * HDIM + col] = x2v[8 + i];
    }
  }
  __syncthreads();

  // ---- norm2 in place on s_x2 ----
  {
    const int row = tid >> 3;
    const int sub = tid & 7;
    float v[16];
    float sum = 0.f, sq = 0.f;
#pragma unroll
    for (int j = 0; j < 16; ++j) {
      float x = s_x2[row * HDIM + sub * 16 + j];
      v[j] = x; sum += x; sq += x * x;
    }
#pragma unroll
    for (int m = 1; m < 8; m <<= 1) { sum += __shfl_xor(sum, m, 32); sq += __shfl_xor(sq, m, 32); }
    float mu   = sum * (1.f / HDIM);
    float rstd = rsqrtf(sq * (1.f / HDIM) - mu * mu + 1e-5f);
#pragma unroll
    for (int j = 0; j < 16; ++j)
      s_x2[row * HDIM + sub * 16 + j] = (v[j] - mu) * rstd;
  }
  __syncthreads();

  // ---- deterministic per-block column sums over 32 rows ----
  if (tid < HDIM) {
    float s = 0.f;
    for (int q = 0; q < 32; ++q) s += s_x2[q * HDIM + tid];
    partials[(size_t)blockIdx.x * HDIM + tid] = s;
  }
}

// ---------------- final reduction: mean over N ----------------
__global__ void k_reduce(const float* __restrict__ partials, float* __restrict__ out, int nblocks) {
  int t = threadIdx.x;  // 128
  float s = 0.f;
  for (int b = 0; b < nblocks; ++b) s += partials[(size_t)b * HDIM + t];
  out[t] = s * (1.0f / (float)NNODES);
}

extern "C" void kernel_launch(void* const* d_in, const int* in_sizes, int n_in,
                              void* d_out, int out_size, void* d_ws, size_t ws_size,
                              hipStream_t stream) {
  const float* hV  = (const float*)d_in[0];
  const float* hE  = (const float*)d_in[1];
  const float* W1  = (const float*)d_in[2];
  const float* b1  = (const float*)d_in[3];
  const float* W2  = (const float*)d_in[4];
  const float* b2  = (const float*)d_in[5];
  const float* W3  = (const float*)d_in[6];
  const float* b3  = (const float*)d_in[7];
  const float* Wd1 = (const float*)d_in[8];
  const float* bd1 = (const float*)d_in[9];
  const float* Wd2 = (const float*)d_in[10];
  const float* bd2 = (const float*)d_in[11];
  float* out = (float*)d_out;

  char* ws = (char*)d_ws;
  _Float16* w1h   = (_Float16*)(ws + 0);         // 384*128 halfs
  _Float16* w2h   = (_Float16*)(ws + 98304);     // 128*128
  _Float16* w3h   = (_Float16*)(ws + 131072);    // 128*128
  _Float16* wd1h  = (_Float16*)(ws + 163840);    // 128*512
  _Float16* wd2h  = (_Float16*)(ws + 294912);    // 512*128
  float* v1       = (float*)(ws + 425984);       // 20000*128 f32
  float* dh       = (float*)(ws + 10665984);     // 20000*128 f32
  float* partials = (float*)(ws + 20905984);     // 625*128 f32

  k_cvt<<<(49152 + 255) / 256, 256, 0, stream>>>(W1, w1h, 49152);
  k_cvt<<<(16384 + 255) / 256, 256, 0, stream>>>(W2, w2h, 16384);
  k_cvt<<<(16384 + 255) / 256, 256, 0, stream>>>(W3, w3h, 16384);
  k_cvt<<<(65536 + 255) / 256, 256, 0, stream>>>(Wd1, wd1h, 65536);
  k_cvt<<<(65536 + 255) / 256, 256, 0, stream>>>(Wd2, wd2h, 65536);

  k_v1<<<NNODES / 16, 256, 0, stream>>>(hV, w1h, b1, v1);

  const int gridB = 2000;
  k_edge_mlp<<<gridB, 256, 0, stream>>>(hE, v1, w1h + 128 * HDIM, w2h, w3h, b2, b3, dh,
                                        NNODES, gridB);

  k_ffn<<<NNODES / 32, 256, 0, stream>>>(hV, dh, wd1h, bd1, wd2h, bd2, partials);
  k_reduce<<<1, HDIM, 0, stream>>>(partials, out, NNODES / 32);
}
